// HAN_7035156431097
// MI455X (gfx1250) — matrix operand order repaired and checked by reading, not compiled, not run on hardware
//
#include <hip/hip_runtime.h>

// ---------------------------------------------------------------------------
// HAN forward for MI455X (gfx1250), wave32 + WMMA f32<=f16 + TDM A-staging.
// Sizes from reference: NS=640 W=64 BS=16 NSENT=40 EMB=512 H=256 NMEN=32 NCAND=32
// ---------------------------------------------------------------------------

typedef __attribute__((ext_vector_type(16))) _Float16 v16h;
typedef __attribute__((ext_vector_type(8)))  float    v8f;
typedef __attribute__((ext_vector_type(4)))  unsigned v4u;
typedef __attribute__((ext_vector_type(8)))  int      v8i;
typedef __attribute__((ext_vector_type(4)))  int      v4i;

#define NS 640
#define W 64
#define BS 16
#define NSENT 40
#define EMB 512
#define H 256
#define O3H 768
#define NMEN 32
#define NCAND 32
#define MREP 2   // m-tiles per workgroup in the GEMM kernels

// ---- WMMA fragment layouts (ISA 7.12.2, wave32) ---------------------------
// A (16x32 f16): lane L holds row m=L%16; g=L/16; half e:
//   e in [0,8):  k = 32*kt + 8*g + e
//   e in [8,16): k = 32*kt + 16 + 8*g + (e-8)
// B (32x16 f16): lane L holds col n=L%16; g=L/16; half e: same k mapping as A
// C/D (16x16 f32): lane L col n=L%16; g=L/16; reg r: m = r + 8*g

__device__ __forceinline__ v8f wmma16(v16h a, v16h b, v8f c) {
    return __builtin_amdgcn_wmma_f32_16x16x32_f16(false, a, false, b,
                                                  (short)0, c, false, false);
}

// ---- TDM: 1-D global->LDS DMA (D# per ISA ch.8 of async_tensor doc) -------
__device__ __forceinline__ void tdm_load_1d(void* lds_dst, const void* gsrc,
                                            unsigned bytes) {
#if __has_builtin(__builtin_amdgcn_tensor_load_to_lds)
    unsigned lds = (unsigned)(unsigned long long)lds_dst;  // LDS addr = low 32b
    unsigned long long ga = (unsigned long long)gsrc;
    unsigned n8 = bytes >> 3;                 // 8-byte elements (data_size=3)
    v4u g0;
    g0[0] = 1u;                               // count=1, user mode
    g0[1] = lds;                              // lds_addr
    g0[2] = (unsigned)ga;                     // global_addr[31:0]
    g0[3] = (unsigned)(ga >> 32) | 0x80000000u; // global_addr[56:32] | type=2
    v8i g1;
    g1[0] = (int)(3u << 16);                  // wg_mask=0, data_size=8B
    g1[1] = (int)((n8 & 0xffffu) << 16);      // tensor_dim0[15:0]
    g1[2] = (int)((n8 >> 16) & 0xffffu) | (1 << 16); // td0[31:16], tensor_dim1=1
    g1[3] = (int)((n8 & 0xffffu) << 16);      // tile_dim0 = n8
    g1[4] = 0;                                // tile_dim1=0, tile_dim2=0
    g1[5] = (int)n8;                          // tensor_dim0_stride[31:0]
    g1[6] = 0;
    g1[7] = 0;
    v4i z4 = {0, 0, 0, 0};
#if defined(__clang_major__) && __clang_major__ >= 23
    v8i z8 = {0, 0, 0, 0, 0, 0, 0, 0};
    __builtin_amdgcn_tensor_load_to_lds(g0, g1, z4, z4, z8, 0);
#else
    __builtin_amdgcn_tensor_load_to_lds(g0, g1, z4, z4, 0);
#endif
#if __has_builtin(__builtin_amdgcn_s_wait_tensorcnt)
    __builtin_amdgcn_s_wait_tensorcnt((short)0);
#endif
#else
    (void)lds_dst; (void)gsrc; (void)bytes;
#endif
}

// Stage nfrag v16h fragments (contiguous) from global into LDS, block-wide.
__device__ __forceinline__ void stage_A_lds(_Float16* Alds, const v16h* Asrc,
                                            int nfrag) {
#if __has_builtin(__builtin_amdgcn_tensor_load_to_lds)
    if (threadIdx.x < 32) {
        tdm_load_1d(Alds, (const void*)Asrc, (unsigned)nfrag * 32u);
    }
#else
    v16h* dst = (v16h*)Alds;
    for (int i = threadIdx.x; i < nfrag; i += 256) dst[i] = Asrc[i];
#endif
    __syncthreads();
}

// ---------------------------------------------------------------------------
__global__ void k_zero(float* p, int n) {
    int i = blockIdx.x * 256 + threadIdx.x;
    if (i < n) p[i] = 0.f;
}

// W is (N x K) row-major (= Wih[o][k]); build B fragments: B[k][n] = W[n][k]
__global__ void k_swizzle_B(const float* __restrict__ Wm, v16h* __restrict__ out,
                            int Ktiles, int K, int total) {
    int idx = blockIdx.x * 256 + threadIdx.x;
    if (idx >= total) return;
    int lane = idx & 31;
    int tmp  = idx >> 5;
    int kt   = tmp % Ktiles;
    int nt   = tmp / Ktiles;
    int n = nt * 16 + (lane & 15);
    int g = lane >> 4;
    const float* row = Wm + (size_t)n * K + kt * 32;
    v16h h;
#pragma unroll
    for (int e = 0; e < 8; ++e) {
        h[e]     = (_Float16)row[8 * g + e];
        h[8 + e] = (_Float16)row[16 + 8 * g + e];
    }
    out[idx] = h;
}

// X is (M x K) row-major f32; build A fragments
__global__ void k_swizzle_A(const float* __restrict__ X, v16h* __restrict__ out,
                            int Ktiles, int K, int total) {
    int idx = blockIdx.x * 256 + threadIdx.x;
    if (idx >= total) return;
    int lane = idx & 31;
    int tmp  = idx >> 5;
    int kt   = tmp % Ktiles;
    int mt   = tmp / Ktiles;
    int m = mt * 16 + (lane & 15);
    int g = lane >> 4;
    const float* row = X + (size_t)m * K + kt * 32;
    v16h h;
#pragma unroll
    for (int e = 0; e < 8; ++e) {
        h[e]     = (_Float16)row[8 * g + e];
        h[8 + e] = (_Float16)row[16 + 8 * g + e];
    }
    out[idx] = h;
}

// Fused token-embedding gather -> A fragments (m = s*64+t, K=512)
__global__ void k_gatherA_words(const int* __restrict__ bdoc,
                                const float* __restrict__ emb,
                                v16h* __restrict__ out, int Ktiles, int total) {
    int idx = blockIdx.x * 256 + threadIdx.x;
    if (idx >= total) return;
    int lane = idx & 31;
    int tmp  = idx >> 5;
    int kt   = tmp % Ktiles;
    int mt   = tmp / Ktiles;
    int m = mt * 16 + (lane & 15);
    int g = lane >> 4;
    int tok = bdoc[m];
    const float* row = emb + (size_t)tok * EMB + kt * 32;
    v16h h;
#pragma unroll
    for (int e = 0; e < 8; ++e) {
        h[e]     = (_Float16)row[8 * g + e];
        h[8 + e] = (_Float16)row[16 + 8 * g + e];
    }
    out[idx] = h;
}

// Generic swizzled GEMM: out[(m%Tlen)*Bcnt + m/Tlen][n] = A@B + bias
// grid.x = Mtiles/MREP, grid.y = Ntiles/8, 256 threads (8 waves).
// A block staged via TDM into LDS; B streamed from global/L2 with prefetch.
__global__ void k_gemm(const v16h* __restrict__ A, const v16h* __restrict__ B,
                       const float* __restrict__ bias, float* __restrict__ out,
                       int Ktiles, int N, int Tlen, int Bcnt) {
    __shared__ __align__(32) _Float16 Alds[MREP * 16 * 32 * 16];  // 32 KB max
    int lane = threadIdx.x & 31;
    int wave = threadIdx.x >> 5;
    int mt0 = blockIdx.x * MREP;
    int nt  = blockIdx.y * 8 + wave;

    stage_A_lds(Alds, A + (size_t)mt0 * Ktiles * 32, MREP * Ktiles * 32);

    const v16h* Av = (const v16h*)Alds;
    const v16h* Bp = B + (size_t)nt * Ktiles * 32 + lane;
    v8f acc[MREP];
    acc[0] = {};
    acc[1] = {};
#pragma unroll 4
    for (int kt = 0; kt < Ktiles; ++kt) {
        if (kt + 2 < Ktiles)
            __builtin_prefetch((const void*)(Bp + (size_t)(kt + 2) * 32), 0, 1);
        v16h b  = Bp[(size_t)kt * 32];
        v16h a0 = Av[kt * 32 + lane];
        v16h a1 = Av[(Ktiles + kt) * 32 + lane];
        acc[0] = wmma16(a0, b, acc[0]);
        acc[1] = wmma16(a1, b, acc[1]);
    }
    int g = lane >> 4;
    int n = nt * 16 + (lane & 15);
    float bn = bias[n];
#pragma unroll
    for (int rep = 0; rep < MREP; ++rep) {
#pragma unroll
        for (int r = 0; r < 8; ++r) {
            int m = (mt0 + rep) * 16 + r + 8 * g;
            int t = m % Tlen;
            int b = m / Tlen;
            out[((size_t)t * Bcnt + b) * (size_t)N + n] = acc[rep][r] + bn;
        }
    }
}

// Attention linear: att[m] += sum_n tanh((A@B)[m][n] + lin_b[n]) * attw[n]
__global__ void k_att_gemm(const v16h* __restrict__ A, const v16h* __restrict__ B,
                           const float* __restrict__ lin_b, const float* __restrict__ attw,
                           float* __restrict__ att, int Ktiles) {
    __shared__ __align__(32) _Float16 Alds[MREP * 16 * 32 * 16];
    int lane = threadIdx.x & 31;
    int wave = threadIdx.x >> 5;
    int mt0 = blockIdx.x * MREP;
    int nt  = blockIdx.y * 8 + wave;

    stage_A_lds(Alds, A + (size_t)mt0 * Ktiles * 32, MREP * Ktiles * 32);

    const v16h* Av = (const v16h*)Alds;
    const v16h* Bp = B + (size_t)nt * Ktiles * 32 + lane;
    v8f acc[MREP];
    acc[0] = {};
    acc[1] = {};
#pragma unroll 4
    for (int kt = 0; kt < Ktiles; ++kt) {
        v16h b  = Bp[(size_t)kt * 32];
        v16h a0 = Av[kt * 32 + lane];
        v16h a1 = Av[(Ktiles + kt) * 32 + lane];
        acc[0] = wmma16(a0, b, acc[0]);
        acc[1] = wmma16(a1, b, acc[1]);
    }
    int g = lane >> 4;
    int n = nt * 16 + (lane & 15);
    float lb = lin_b[n], aw = attw[n];
#pragma unroll
    for (int rep = 0; rep < MREP; ++rep) {
#pragma unroll
        for (int r = 0; r < 8; ++r) {
            int m = (mt0 + rep) * 16 + r + 8 * g;
            atomicAdd(&att[m], tanhf(acc[rep][r] + lb) * aw);
        }
    }
}

// Bidirectional GRU scan. One block handles 16 batch rows across all T steps.
// h kept in LDS as f16 already in A-fragment layout. grid(Bc/16, 2=dir).
__global__ void k_gru_scan(const float* __restrict__ xp_f, const float* __restrict__ xp_b,
                           const v16h* __restrict__ WBf, const v16h* __restrict__ WBb,
                           const float* __restrict__ bhhf, const float* __restrict__ bhhb,
                           const int* __restrict__ lens, float* __restrict__ enc,
                           int T, int Bc) {
    __shared__ __align__(32) _Float16 hA[8][32][16];  // 8 KB : h as A-fragments
    __shared__ float g_s[16][O3H];                    // 48 KB: h@Whh^T + bhh
    int tid  = threadIdx.x;
    int lane = tid & 31;
    int wave = tid >> 5;
    int dir  = blockIdx.y;
    const float* xp  = dir ? xp_b : xp_f;
    const v16h* WB   = dir ? WBb : WBf;
    const float* bhh = dir ? bhhb : bhhf;

    for (int i = tid; i < 8 * 32 * 16; i += 256)
        ((_Float16*)hA)[i] = (_Float16)0.f;
    __syncthreads();

    int row = tid >> 4;                 // 0..15
    int jb  = tid & 15;
    int bg  = blockIdx.x * 16 + row;    // batch element this thread's gates own
    int len = lens[bg];

    for (int step = 0; step < T; ++step) {
        int t = dir ? (T - 1 - step) : step;
        // --- g = h @ Whh^T + bhh via WMMA (8 waves x 6 n-tiles) ---
#pragma unroll
        for (int q = 0; q < 6; ++q) {
            int nt = wave * 6 + q;
            v8f acc = {};
#pragma unroll
            for (int kt = 0; kt < 8; ++kt) {
                v16h a = *(const v16h*)&hA[kt][lane][0];
                v16h b = WB[((size_t)nt * 8 + kt) * 32 + lane];
                acc = wmma16(a, b, acc);
            }
            int gg = lane >> 4;
            int n  = nt * 16 + (lane & 15);
            float bv = bhh[n];
#pragma unroll
            for (int r = 0; r < 8; ++r)
                g_s[r + 8 * gg][n] = acc[r] + bv;
        }
        __syncthreads();
        // --- gates (f32 VALU), update h, emit ys ---
        bool valid = (t < len);
        size_t xbase = ((size_t)t * Bc + bg) * (size_t)O3H;
        size_t obase = ((size_t)t * Bc + bg) * (size_t)(2 * H) + (size_t)dir * H;
#pragma unroll
        for (int i = 0; i < 16; ++i) {
            int j  = jb + 16 * i;           // 0..255
            int kt = j >> 5;
            int kp = j & 31;
            int gA = (kp >> 3) & 1;
            int e  = kp - 8 * gA - ((kp >= 16) ? 8 : 0);
            int L  = row + 16 * gA;
            float h_old = (float)hA[kt][L][e];
            float xr = xp[xbase + j], xz = xp[xbase + H + j], xn = xp[xbase + 2 * H + j];
            float hr = g_s[row][j], hz = g_s[row][H + j], hn = g_s[row][2 * H + j];
            float rg = 1.f / (1.f + expf(-(xr + hr)));
            float zg = 1.f / (1.f + expf(-(xz + hz)));
            float ng = tanhf(xn + rg * hn);
            float hnew = valid ? ((1.f - zg) * ng + zg * h_old) : h_old;
            hA[kt][L][e] = (_Float16)hnew;
            enc[obase + j] = valid ? hnew : 0.f;
        }
        __syncthreads();
    }
}

// Softmax-style attention pool: outv[b] = sum_t (e_t/denom) * enc[t][b][:]
__global__ void k_att_reduce(const float* __restrict__ att, const float* __restrict__ enc,
                             const int* __restrict__ lens, float* __restrict__ outv,
                             int T, int Bc) {
    __shared__ float earr[64];
    __shared__ float red[256];
    int b = blockIdx.x, tid = threadIdx.x;
    int len = lens[b];
    if (tid < T) earr[tid] = (tid < len) ? expf(att[(size_t)tid * Bc + b]) : 0.f;
    __syncthreads();
    float part = (tid < T) ? earr[tid] : 0.f;
    red[tid] = part;
    __syncthreads();
    for (int s = 128; s > 0; s >>= 1) {
        if (tid < s) red[tid] += red[tid + s];
        __syncthreads();
    }
    float denom = red[0] + 1e-4f;
    for (int c = tid; c < 2 * H; c += 256) {
        float acc = 0.f;
        for (int t = 0; t < T; ++t)
            acc += earr[t] * enc[((size_t)t * Bc + b) * (size_t)(2 * H) + c];
        outv[(size_t)b * (2 * H) + c] = acc / denom;
    }
}

__global__ void k_gather_sent(const int* __restrict__ sent_order,
                              const float* __restrict__ sent_embs,
                              float* __restrict__ each_sent, int n) {
    int g = blockIdx.x * 256 + threadIdx.x;
    if (g >= n) return;
    int c = g & 511;
    int m = g >> 9;
    int so = sent_order[m];
    each_sent[g] = (so == 0) ? 0.f : sent_embs[(size_t)(so - 1) * 512 + c];
}

__device__ __forceinline__ float blk_reduce(float v, float* red) {
    int tid = threadIdx.x;
    red[tid] = v;
    __syncthreads();
    for (int s = 128; s > 0; s >>= 1) {
        if (tid < s) red[tid] += red[tid + s];
        __syncthreads();
    }
    float r = red[0];
    __syncthreads();
    return r;
}

// Cosine sims + MLP. One block per (doc, mention).
__global__ void k_final(const float* __restrict__ each_sent, const float* __restrict__ doc_embs,
                        const int* __restrict__ men_sent_idx, const int* __restrict__ can_ent_idx,
                        const float* __restrict__ cand_mask, const float* __restrict__ embed_ent,
                        const float* __restrict__ mlp_w, const float* __restrict__ mlp_b,
                        float* __restrict__ out) {
    __shared__ float sv[512], dv[512], red[256], ssl[NCAND], dsl[NCAND];
    int tid = threadIdx.x;
    int d = blockIdx.x >> 5, men = blockIdx.x & 31;
    int j = men_sent_idx[d * NMEN + men];
    for (int c = tid; c < 512; c += 256) {
        sv[c] = each_sent[(size_t)(d * NSENT + j) * 512 + c];
        dv[c] = doc_embs[(size_t)d * 512 + c];
    }
    __syncthreads();
    float ps = 0.f, pd = 0.f;
    for (int c = tid; c < 512; c += 256) { ps += sv[c] * sv[c]; pd += dv[c] * dv[c]; }
    float na = sqrtf(blk_reduce(ps, red));
    float nd = sqrtf(blk_reduce(pd, red));
    for (int cand = 0; cand < NCAND; ++cand) {
        int ei = can_ent_idx[((size_t)(d * NMEN + men)) * NCAND + cand];
        const float* ev = embed_ent + (size_t)ei * EMB;
        float as = 0.f, ad = 0.f, nn = 0.f;
        for (int c = tid; c < 512; c += 256) {
            float e = ev[c];
            as += sv[c] * e; ad += dv[c] * e; nn += e * e;
        }
        float dots = blk_reduce(as, red);
        float dotd = blk_reduce(ad, red);
        float ne2  = blk_reduce(nn, red);
        if (tid == 0) {
            float ne  = sqrtf(ne2);
            float msk = cand_mask[((size_t)(d * NMEN + men)) * NCAND + cand];
            ssl[cand] = msk * dots / fmaxf(na * ne, 1e-8f);
            dsl[cand] = msk * dotd / fmaxf(nd * ne, 1e-8f);
        }
        __syncthreads();
    }
    if (tid < NCAND) {
        float acc = mlp_b[tid];
        for (int c = 0; c < NCAND; ++c)
            acc += ssl[c] * mlp_w[tid * 64 + c] + dsl[c] * mlp_w[tid * 64 + 32 + c];
        float m = (ssl[tid] != 0.f) ? 1.f : 0.f;
        out[(size_t)blockIdx.x * NCAND + tid] = acc * m;
    }
}

// ---------------------------------------------------------------------------
extern "C" void kernel_launch(void* const* d_in, const int* in_sizes, int n_in,
                              void* d_out, int out_size, void* d_ws, size_t ws_size,
                              hipStream_t stream) {
    (void)in_sizes; (void)n_in; (void)out_size; (void)ws_size;
    const int*   batch_doc    = (const int*)d_in[0];
    const int*   sent_order   = (const int*)d_in[1];
    const int*   length_sent  = (const int*)d_in[2];
    const int*   length_doc   = (const int*)d_in[3];
    const int*   men_sent_idx = (const int*)d_in[4];
    const int*   can_ent_idx  = (const int*)d_in[5];
    const float* cand_mask    = (const float*)d_in[6];
    const float* embed_doc    = (const float*)d_in[7];
    const float* embed_ent    = (const float*)d_in[8];
    const float* wWihf = (const float*)d_in[9];
    const float* wWhhf = (const float*)d_in[10];
    const float* wbihf = (const float*)d_in[11];
    const float* wbhhf = (const float*)d_in[12];
    const float* wWihb = (const float*)d_in[13];
    const float* wWhhb = (const float*)d_in[14];
    const float* wbihb = (const float*)d_in[15];
    const float* wbhhb = (const float*)d_in[16];
    const float* wlinw = (const float*)d_in[17];
    const float* wlinb = (const float*)d_in[18];
    const float* wattw = (const float*)d_in[19];
    const float* sWihf = (const float*)d_in[20];
    const float* sWhhf = (const float*)d_in[21];
    const float* sbihf = (const float*)d_in[22];
    const float* sbhhf = (const float*)d_in[23];
    const float* sWihb = (const float*)d_in[24];
    const float* sWhhb = (const float*)d_in[25];
    const float* sbihb = (const float*)d_in[26];
    const float* sbhhb = (const float*)d_in[27];
    const float* slinw = (const float*)d_in[28];
    const float* slinb = (const float*)d_in[29];
    const float* sattw = (const float*)d_in[30];
    const float* mlpw  = (const float*)d_in[31];
    const float* mlpb  = (const float*)d_in[32];

    char* base = (char*)d_ws;
    size_t cur = 0;
    auto alloc = [&](size_t bytes) -> char* {
        cur = (cur + 255) & ~(size_t)255;
        char* p = base + cur;
        cur += bytes;
        return p;
    };

    const int M1 = NS * W;                 // 40960
    const int M1t = M1 / 16;               // 2560
    const int M2 = BS * NSENT;             // 640
    const int M2t = M2 / 16;               // 40

    v16h* A1      = (v16h*)alloc((size_t)M1t * 16 * 32 * 32);      // 41.9 MB
    v16h* B1f     = (v16h*)alloc((size_t)48 * 16 * 32 * 32);
    v16h* B1b     = (v16h*)alloc((size_t)48 * 16 * 32 * 32);
    float* xp1f   = (float*)alloc((size_t)W * NS * O3H * 4);       // 125.8 MB
    float* xp1b   = (float*)alloc((size_t)W * NS * O3H * 4);       // 125.8 MB
    v16h* Whh1fB  = (v16h*)alloc((size_t)48 * 8 * 32 * 32);
    v16h* Whh1bB  = (v16h*)alloc((size_t)48 * 8 * 32 * 32);
    float* enc1   = (float*)alloc((size_t)W * NS * 2 * H * 4);     // 83.9 MB
    v16h* enc1A   = (v16h*)alloc((size_t)M1t * 16 * 32 * 32);      // 41.9 MB
    v16h* linw1B  = (v16h*)alloc((size_t)32 * 16 * 32 * 32);
    float* att1   = (float*)alloc((size_t)M1 * 4);
    float* sent_e = (float*)alloc((size_t)NS * 512 * 4);
    float* each_s = (float*)alloc((size_t)BS * NSENT * 512 * 4);
    v16h* A2      = (v16h*)alloc((size_t)M2t * 16 * 32 * 32);
    v16h* B2f     = (v16h*)alloc((size_t)48 * 16 * 32 * 32);
    v16h* B2b     = (v16h*)alloc((size_t)48 * 16 * 32 * 32);
    float* xp2f   = (float*)alloc((size_t)NSENT * BS * O3H * 4);
    float* xp2b   = (float*)alloc((size_t)NSENT * BS * O3H * 4);
    v16h* Whh2fB  = (v16h*)alloc((size_t)48 * 8 * 32 * 32);
    v16h* Whh2bB  = (v16h*)alloc((size_t)48 * 8 * 32 * 32);
    float* enc2   = (float*)alloc((size_t)NSENT * BS * 2 * H * 4);
    v16h* enc2A   = (v16h*)alloc((size_t)M2t * 16 * 32 * 32);
    v16h* linw2B  = (v16h*)alloc((size_t)32 * 16 * 32 * 32);
    float* att2   = (float*)alloc((size_t)M2 * 4);
    float* doc_e  = (float*)alloc((size_t)BS * 512 * 4);

    // ---- weight swizzles (f32 -> f16 WMMA B fragments) ----
    int tIh = 48 * 16 * 32, tHh = 48 * 8 * 32, tLn = 32 * 16 * 32;
    k_swizzle_B<<<(tIh + 255) / 256, 256, 0, stream>>>(wWihf, B1f, 16, 512, tIh);
    k_swizzle_B<<<(tIh + 255) / 256, 256, 0, stream>>>(wWihb, B1b, 16, 512, tIh);
    k_swizzle_B<<<(tHh + 255) / 256, 256, 0, stream>>>(wWhhf, Whh1fB, 8, 256, tHh);
    k_swizzle_B<<<(tHh + 255) / 256, 256, 0, stream>>>(wWhhb, Whh1bB, 8, 256, tHh);
    k_swizzle_B<<<(tLn + 255) / 256, 256, 0, stream>>>(wlinw, linw1B, 16, 512, tLn);
    k_swizzle_B<<<(tIh + 255) / 256, 256, 0, stream>>>(sWihf, B2f, 16, 512, tIh);
    k_swizzle_B<<<(tIh + 255) / 256, 256, 0, stream>>>(sWihb, B2b, 16, 512, tIh);
    k_swizzle_B<<<(tHh + 255) / 256, 256, 0, stream>>>(sWhhf, Whh2fB, 8, 256, tHh);
    k_swizzle_B<<<(tHh + 255) / 256, 256, 0, stream>>>(sWhhb, Whh2bB, 8, 256, tHh);
    k_swizzle_B<<<(tLn + 255) / 256, 256, 0, stream>>>(slinw, linw2B, 16, 512, tLn);

    // ---- level 1: words ----
    int tA1 = M1t * 16 * 32;
    k_gatherA_words<<<(tA1 + 255) / 256, 256, 0, stream>>>(batch_doc, embed_doc, A1, 16, tA1);
    k_gemm<<<dim3(M1t / MREP, 6), 256, 0, stream>>>(A1, B1f, wbihf, xp1f, 16, O3H, W, NS);
    k_gemm<<<dim3(M1t / MREP, 6), 256, 0, stream>>>(A1, B1b, wbihb, xp1b, 16, O3H, W, NS);
    k_gru_scan<<<dim3(NS / 16, 2), 256, 0, stream>>>(xp1f, xp1b, Whh1fB, Whh1bB,
                                                     wbhhf, wbhhb, length_sent, enc1, W, NS);
    k_swizzle_A<<<(tA1 + 255) / 256, 256, 0, stream>>>(enc1, enc1A, 16, 512, tA1);
    k_zero<<<(M1 + 255) / 256, 256, 0, stream>>>(att1, M1);
    k_att_gemm<<<dim3(M1t / MREP, 4), 256, 0, stream>>>(enc1A, linw1B, wlinb, wattw, att1, 16);
    k_att_reduce<<<NS, 256, 0, stream>>>(att1, enc1, length_sent, sent_e, W, NS);

    // ---- level 2: sentences ----
    int nES = BS * NSENT * 512;
    k_gather_sent<<<(nES + 255) / 256, 256, 0, stream>>>(sent_order, sent_e, each_s, nES);
    int tA2 = M2t * 16 * 32;
    k_swizzle_A<<<(tA2 + 255) / 256, 256, 0, stream>>>(each_s, A2, 16, 512, tA2);
    k_gemm<<<dim3(M2t / MREP, 6), 256, 0, stream>>>(A2, B2f, sbihf, xp2f, 16, O3H, NSENT, BS);
    k_gemm<<<dim3(M2t / MREP, 6), 256, 0, stream>>>(A2, B2b, sbihb, xp2b, 16, O3H, NSENT, BS);
    k_gru_scan<<<dim3(1, 2), 256, 0, stream>>>(xp2f, xp2b, Whh2fB, Whh2bB,
                                               sbhhf, sbhhb, length_doc, enc2, NSENT, BS);
    k_swizzle_A<<<(tA2 + 255) / 256, 256, 0, stream>>>(enc2, enc2A, 16, 512, tA2);
    k_zero<<<(M2 + 255) / 256, 256, 0, stream>>>(att2, M2);
    k_att_gemm<<<dim3(M2t / MREP, 4), 256, 0, stream>>>(enc2A, linw2B, slinb, sattw, att2, 16);
    k_att_reduce<<<BS, 256, 0, stream>>>(att2, enc2, length_doc, doc_e, NSENT, BS);

    // ---- cosine sims + MLP ----
    k_final<<<BS * NMEN, 256, 0, stream>>>(each_s, doc_e, men_sent_idx, can_ent_idx,
                                           cand_mask, embed_ent, mlpw, mlpb, (float*)d_out);
}
